// ExponentialConcordanceLoss_75634374083215
// MI455X (gfx1250) — compile-verified, matrix operand order checked by reading, not run on hardware
//
#include <hip/hip_runtime.h>
#include <math.h>

typedef __attribute__((ext_vector_type(16))) _Float16 v16h;
typedef __attribute__((ext_vector_type(8)))  float    v8f;
typedef __attribute__((ext_vector_type(8)))  unsigned v8u;

#define NN 8192

// ws layout (as float words):
//   [0] loss_sum   [1] count_sum   [2..7] pad
//   [8      .. 8+NN)   aArr  : finite_i ? exp(s_i) : 0        (f32 row factor, loss)
//   [8+NN   .. 8+2NN)  fwArr : finite_i ? 1 : 0               (f32 row factor, count)
//   [8+2NN  .. 8+3NN)  tArr  : raw t_j                        (f32, NaN compares false)
//   [8+3NN  .. 8+4NN)  bcArr : packed {hi: cq=ev?1:0, lo: bq=ev?exp(-s_j):0} as 2x f16

__global__ void ecl_prep_kernel(const float* __restrict__ preds,
                                const float* __restrict__ targets,
                                float* __restrict__ ws) {
    int j = blockIdx.x * blockDim.x + threadIdx.x;
    if (j == 0) { ws[0] = 0.0f; ws[1] = 0.0f; }
    if (j >= NN) return;

    float s = preds[j];
    float t = targets[2 * j];
    float e = targets[2 * j + 1];
    bool fin = __builtin_isfinite(t) && __builtin_isfinite(s);
    bool ev  = fin && (e != 0.0f);

    float* aArr  = ws + 8;
    float* fwArr = ws + 8 + NN;
    float* tArr  = ws + 8 + 2 * NN;
    unsigned* bcArr = (unsigned*)(ws + 8 + 3 * NN);

    aArr[j]  = fin ? expf(s) : 0.0f;
    fwArr[j] = fin ? 1.0f : 0.0f;
    tArr[j]  = t;  // raw: NaN rows/cols are killed by zero weights; NaN compares false

    _Float16 bh = ev ? (_Float16)expf(-s) : (_Float16)0.0f;
    _Float16 ch = ev ? (_Float16)1.0f     : (_Float16)0.0f;
    unsigned bu = (unsigned)__builtin_bit_cast(unsigned short, bh);
    unsigned cu = (unsigned)__builtin_bit_cast(unsigned short, ch);
    bcArr[j] = bu | (cu << 16);
}

// One wave per 16-row i-tile; 8 waves per block; sweep all j in K=32 chunks with WMMA.
__global__ __launch_bounds__(256) void ecl_pair_kernel(float* __restrict__ ws) {
    __shared__ float    sT[NN];    // 32 KB
    __shared__ unsigned sBC[NN];   // 32 KB

    const float*    tArr  = ws + 8 + 2 * NN;
    const unsigned* bcArr = (const unsigned*)(ws + 8 + 3 * NN);

    for (int idx = threadIdx.x; idx < NN; idx += blockDim.x) {
        sT[idx]  = tArr[idx];
        sBC[idx] = bcArr[idx];
    }
    __syncthreads();

    const int lane  = threadIdx.x & 31;
    const int wave  = threadIdx.x >> 5;
    const int ibase = (blockIdx.x * 8 + wave) * 16;
    const int m     = lane & 15;         // A-matrix row owned by this lane
    const int h     = lane >> 4;         // lane half selects K sub-range

    const float trow = sT[ibase + m];

    v16h ones;
    #pragma unroll
    for (int q = 0; q < 16; ++q) ones[q] = (_Float16)1.0f;

    v8f accB = {};  // row sums of mask * exp(-s_j)
    v8f accC = {};  // row sums of mask * 1

    for (int jb = 0; jb < NN; jb += 32) {
        v8u ab, ac;
        // ISA 16-bit A(16x32) layout: VGPR p (=4g+q) holds K = ((p&4)<<2) + h*8 + (p&3)*2
        // -> per lane, two contiguous 8-element runs: [h*8, h*8+8) and [16+h*8, 16+h*8+8)
        #pragma unroll
        for (int g = 0; g < 2; ++g) {
            const int jrun = jb + g * 16 + h * 8;

            // Unconditional loads (coalesce to ds_load_b128), then bitwise masking:
            // avoid ternaries on loaded values so the compiler can't predicate loads.
            float    tv[8];
            unsigned bv[8];
            #pragma unroll
            for (int q = 0; q < 8; ++q) { tv[q] = sT[jrun + q]; }
            #pragma unroll
            for (int q = 0; q < 8; ++q) { bv[q] = sBC[jrun + q]; }

            #pragma unroll
            for (int q = 0; q < 4; ++q) {
                const unsigned m0 = 0u - (unsigned)(tv[2 * q]     < trow);
                const unsigned m1 = 0u - (unsigned)(tv[2 * q + 1] < trow);
                const unsigned b0 = bv[2 * q]     & m0;
                const unsigned b1 = bv[2 * q + 1] & m1;
                // byte-permute packs: ab = {b1.lo16, b0.lo16}, ac = {b1.hi16, b0.hi16}
                ab[g * 4 + q] = __builtin_amdgcn_perm(b1, b0, 0x05040100u);
                ac[g * 4 + q] = __builtin_amdgcn_perm(b1, b0, 0x07060302u);
            }
        }
        v16h Ab = __builtin_bit_cast(v16h, ab);
        v16h Ac = __builtin_bit_cast(v16h, ac);
        // D[m][n] += sum_k A[m][k] * 1  -> row sums replicated across all 16 columns
        accB = __builtin_amdgcn_wmma_f32_16x16x32_f16(false, Ab, false, ones,
                                                      (short)0, accB, false, false);
        accC = __builtin_amdgcn_wmma_f32_16x16x32_f16(false, Ac, false, ones,
                                                      (short)0, accC, false, false);
    }

    // C/D layout: lane holds column N = lane&15, rows v + 8*h. Column 0 lanes reduce.
    if (m == 0) {
        const float* aArr  = ws + 8;
        const float* fwArr = ws + 8 + NN;
        float lsum = 0.0f, csum = 0.0f;
        #pragma unroll
        for (int v = 0; v < 8; ++v) {
            const int row = ibase + v + 8 * h;
            lsum += accB[v] * aArr[row];    // f32 row factor exp(s_i)
            csum += accC[v] * fwArr[row];   // f32 row factor finite_i
        }
        atomicAdd(&ws[0], lsum);
        atomicAdd(&ws[1], csum);
    }
}

__global__ void ecl_finalize_kernel(const float* __restrict__ ws,
                                    float* __restrict__ out) {
    out[0] = ws[0] / fmaxf(ws[1], 1.0f);
}

extern "C" void kernel_launch(void* const* d_in, const int* in_sizes, int n_in,
                              void* d_out, int out_size, void* d_ws, size_t ws_size,
                              hipStream_t stream) {
    const float* preds   = (const float*)d_in[0];   // (N,)   f32
    const float* targets = (const float*)d_in[1];   // (N,2)  f32 interleaved [dur, evt]
    float* out = (float*)d_out;
    float* ws  = (float*)d_ws;

    ecl_prep_kernel<<<(NN + 255) / 256, 256, 0, stream>>>(preds, targets, ws);
    // 512 i-tiles of 16 rows; 8 waves/block -> 64 blocks
    ecl_pair_kernel<<<64, 256, 0, stream>>>(ws);
    ecl_finalize_kernel<<<1, 1, 0, stream>>>(ws, out);
}
